// GAE_AT_dense_20212116095374
// MI455X (gfx1250) — compile-verified, matrix operand order checked by reading, not run on hardware
//
#include <hip/hip_runtime.h>
#include <hip/hip_bf16.h>
#include <math.h>

// ---------------- problem constants ----------------
constexpr int kN     = 6144;
constexpr int kIN    = 256;
constexpr int kHID   = 64;
constexpr int kHEADS = 4;
constexpr int kLAT   = 32;
constexpr float kALPHA = 0.2f;

typedef __attribute__((ext_vector_type(16))) __bf16 v16bf;
typedef __attribute__((ext_vector_type(8)))  float  v8f;
typedef int v4i __attribute__((vector_size(16)));

#define AS1 __attribute__((address_space(1)))
#define AS3 __attribute__((address_space(3)))

#if defined(__gfx1250__) && __has_builtin(__builtin_amdgcn_global_load_async_to_lds_b128)
#define HAVE_ASYNC_LDS 1
#else
#define HAVE_ASYNC_LDS 0
#endif
#if defined(__gfx1250__) && __has_builtin(__builtin_amdgcn_s_wait_asynccnt)
#define HAVE_WAIT_ASYNC 1
#else
#define HAVE_WAIT_ASYNC 0
#endif

// ---------------- helpers ----------------
__device__ __forceinline__ unsigned short f32_to_bf16_rne(float f) {
    unsigned int u = __float_as_uint(f);
    unsigned int r = u + 0x7fffu + ((u >> 16) & 1u);
    return (unsigned short)(r >> 16);
}
__device__ __forceinline__ __bf16 bfbits(unsigned short b) {
    return __builtin_bit_cast(__bf16, b);
}
// CDNA5 16-bit A/B fragment K index for element i (0..15), given lane-half.
__device__ __forceinline__ int frag_k(int i, int hlf) {
    int v = i >> 1, p = i & 1;
    return (v < 4) ? (hlf * 8 + v * 2 + p) : (16 + hlf * 8 + (v - 4) * 2 + p);
}

// ---------------- elementwise converts ----------------
__global__ __launch_bounds__(256) void f2bf_kernel(const float* __restrict__ in,
                                                   unsigned short* __restrict__ out, int n) {
    int idx = blockIdx.x * 256 + threadIdx.x;
    if (idx < n) out[idx] = f32_to_bf16_rne(in[idx]);
}

// W_heads [H][IN][HID] -> Wc bf16 [IN][H*HID] (heads concatenated along cols)
__global__ __launch_bounds__(256) void pack_wheads_kernel(const float* __restrict__ w,
                                                          unsigned short* __restrict__ out) {
    int idx = blockIdx.x * 256 + threadIdx.x;
    int total = kHEADS * kIN * kHID;
    if (idx >= total) return;
    int h = idx / (kIN * kHID);
    int rem = idx % (kIN * kHID);
    int k = rem / kHID, f = rem % kHID;
    out[(size_t)k * (kHEADS * kHID) + h * kHID + f] = f32_to_bf16_rne(w[idx]);
}

// ---------------- generic bf16 WMMA GEMM: C[M,Nc] = A[M,K] @ B[K,Nc] ----------------
__global__ __launch_bounds__(256) void wmma_gemm_bf16(const unsigned short* __restrict__ A,
                                                      const unsigned short* __restrict__ B,
                                                      float* __restrict__ C,
                                                      int M, int K, int Nc) {
    int wave = (blockIdx.x * blockDim.x + threadIdx.x) >> 5;
    int tilesN = Nc >> 4;
    int totalTiles = (M >> 4) * tilesN;
    if (wave >= totalTiles) return;
    int tm = wave / tilesN, tn = wave % tilesN;
    int lane = threadIdx.x & 31;
    int hlf = lane >> 4, r = lane & 15;
    v8f c = {0.f, 0.f, 0.f, 0.f, 0.f, 0.f, 0.f, 0.f};
    for (int k0 = 0; k0 < K; k0 += 32) {
        v16bf a, b;
#pragma unroll
        for (int i = 0; i < 16; ++i) {
            int kk = frag_k(i, hlf);
            a[i] = bfbits(A[(size_t)(tm * 16 + r) * K + k0 + kk]);
            b[i] = bfbits(B[(size_t)(k0 + kk) * Nc + tn * 16 + r]);
        }
        c = __builtin_amdgcn_wmma_f32_16x16x32_bf16(false, a, false, b, (short)0, c, false, false);
    }
#pragma unroll
    for (int rr = 0; rr < 8; ++rr)
        C[(size_t)(tm * 16 + hlf * 8 + rr) * Nc + tn * 16 + r] = c[rr];
}

// ---------------- attention edge logits: esrc/edst = Wh_head @ a halves ----------------
__global__ __launch_bounds__(256) void edge_logits_kernel(const float* __restrict__ Wh, int ld,
                                                          int F, int nheads,
                                                          const float* __restrict__ a,
                                                          float* __restrict__ esrc,
                                                          float* __restrict__ edst) {
    int idx = blockIdx.x * 256 + threadIdx.x;
    if (idx >= nheads * kN) return;
    int h = idx / kN, n = idx % kN;
    const float* w  = Wh + (size_t)n * ld + h * F;
    const float* ah = a + h * 2 * F;
    float s = 0.f, d = 0.f;
    for (int f = 0; f < F; ++f) {
        float wv = w[f];
        s += wv * ah[f];
        d += wv * ah[F + f];
    }
    esrc[idx] = s;
    edst[idx] = d;
}

// ---------------- streaming masked-softmax row stats (online max/sum-exp) ----------------
__global__ __launch_bounds__(256) void softmax_stats_kernel(const float* __restrict__ Adj,
                                                            const float* __restrict__ esrc,
                                                            const float* __restrict__ edst,
                                                            float* __restrict__ rmax,
                                                            float* __restrict__ rsum) {
    int i = blockIdx.x;
    int t = threadIdx.x;
    float es = esrc[i];
    const float* arow = Adj + (size_t)i * kN;
    float m = -INFINITY, s = 0.f;
    for (int j = t; j < kN; j += 256) {
        if (j + 2048 < kN) __builtin_prefetch(arow + j + 2048, 0, 1);
        float aj = arow[j];
        if (aj > 0.f) {  // A is 0/1; An>0 <=> A>0
            float e = es + edst[j];
            e = (e > 0.f) ? e : kALPHA * e;
            if (e > m) { s = s * __expf(m - e) + 1.f; m = e; }
            else       { s += __expf(e - m); }
        }
    }
    __shared__ float sm[256], ss[256];
    sm[t] = m; ss[t] = s;
    __syncthreads();
    for (int w = 128; w >= 1; w >>= 1) {
        if (t < w) {
            float m1 = sm[t], s1 = ss[t], m2 = sm[t + w], s2 = ss[t + w];
            if (s2 > 0.f) {
                if (s1 <= 0.f) { sm[t] = m2; ss[t] = s2; }
                else {
                    float M = fmaxf(m1, m2);
                    ss[t] = s1 * __expf(m1 - M) + s2 * __expf(m2 - M);
                    sm[t] = M;
                }
            }
        }
        __syncthreads();
    }
    if (t == 0) {
        rmax[i] = (ss[0] > 0.f) ? sm[0] : 0.f;
        rsum[i] = ss[0];
    }
}

// ---------------- fused masked-softmax @ Wh (flash-style, WMMA, async-LDS staged B) -------
// Block = 8 waves, each wave owns one 16-row tile; all waves share the same j schedule,
// so the 32 x (NF*16) bf16 B-tile is staged cooperatively into LDS via async global->LDS.
template <int NF>
__global__ __launch_bounds__(256) void gat_aggregate_kernel(const float* __restrict__ Adj,
                                                            const float* __restrict__ esrc,
                                                            const float* __restrict__ edst,
                                                            const float* __restrict__ rmax,
                                                            const float* __restrict__ rsum,
                                                            const unsigned short* __restrict__ Whb,
                                                            int ldWh, int colOff,
                                                            float* __restrict__ Out, int ldOut) {
    __shared__ unsigned short Bs[32][72];  // row stride 144B: lane-halves hit disjoint banks
    __shared__ float Eds[32];

    int wave = threadIdx.x >> 5;
    int rowTile = blockIdx.x * 8 + wave;   // launch geometry is exact: no early return
    int lane = threadIdx.x & 31;
    int hlf = lane >> 4, r = lane & 15;
    int i = rowTile * 16 + r;

    float rm = rmax[i];
    float rs = rsum[i];
    float inv_rs = (rs > 0.f) ? (1.f / rs) : 0.f;
    float es = esrc[i];
    const float* arow = Adj + (size_t)i * kN;

    v8f zero = {0.f, 0.f, 0.f, 0.f, 0.f, 0.f, 0.f, 0.f};
    v8f c[NF];
#pragma unroll
    for (int f = 0; f < NF; ++f) c[f] = zero;

    // cooperative copy assignment: 16B chunks, NF*2 chunks per 32-row tile row
    const int t = threadIdx.x;
    const int chunksPerRow = NF * 2;
    const int copyRow = t / chunksPerRow;
    const int copyChunk = t % chunksPerRow;
    const bool copyActive = t < 32 * chunksPerRow;

    for (int j0 = 0; j0 < kN; j0 += 32) {
        if (j0 + 512 < kN) __builtin_prefetch(arow + j0 + 512, 0, 1);

        // ---- stage B tile (and edst strip) into LDS ----
        if (copyActive) {
            const unsigned short* g =
                Whb + (size_t)(j0 + copyRow) * ldWh + colOff + copyChunk * 8;
            unsigned short* l = &Bs[copyRow][copyChunk * 8];
#if HAVE_ASYNC_LDS
            __builtin_amdgcn_global_load_async_to_lds_b128((AS1 v4i*)g, (AS3 v4i*)l, 0, 0);
#else
            *(uint4*)l = *(const uint4*)g;
#endif
        }
        if (t < 32) Eds[t] = edst[j0 + t];
#if HAVE_ASYNC_LDS && HAVE_WAIT_ASYNC
        __builtin_amdgcn_s_wait_asynccnt(0);
#endif
        __syncthreads();

        // ---- A fragment: attention probabilities built on the fly ----
        v16bf a;
#pragma unroll
        for (int ii = 0; ii < 16; ++ii) {
            int kk = frag_k(ii, hlf);
            float aj = arow[j0 + kk];
            float prob = 0.f;
            if (aj > 0.f) {
                float e = es + Eds[kk];
                e = (e > 0.f) ? e : kALPHA * e;
                prob = __expf(e - rm) * inv_rs;
            }
            a[ii] = bfbits(f32_to_bf16_rne(prob));
        }
        // ---- B fragments from LDS ----
#pragma unroll
        for (int f = 0; f < NF; ++f) {
            v16bf b;
#pragma unroll
            for (int ii = 0; ii < 16; ++ii) {
                int kk = frag_k(ii, hlf);
                b[ii] = bfbits(Bs[kk][f * 16 + r]);
            }
            c[f] = __builtin_amdgcn_wmma_f32_16x16x32_bf16(false, a, false, b, (short)0,
                                                           c[f], false, false);
        }
        __syncthreads();
    }
#pragma unroll
    for (int f = 0; f < NF; ++f) {
#pragma unroll
        for (int rr = 0; rr < 8; ++rr) {
            float val = c[f][rr];
            val = (val > 0.f) ? val : (__expf(val) - 1.f);  // ELU
            Out[(size_t)(rowTile * 16 + hlf * 8 + rr) * ldOut + colOff + f * 16 + r] = val;
        }
    }
}

// ---------------- final: sigmoid(z @ z^T), K = 32 = one WMMA ----------------
__global__ __launch_bounds__(256) void zzt_sigmoid_kernel(const unsigned short* __restrict__ zb,
                                                          float* __restrict__ out) {
    int wave = (blockIdx.x * blockDim.x + threadIdx.x) >> 5;
    int tiles = kN >> 4;  // 384
    if (wave >= tiles * tiles) return;
    int tm = wave / tiles, tn = wave % tiles;
    int lane = threadIdx.x & 31;
    int hlf = lane >> 4, r = lane & 15;
    v16bf a, b;
#pragma unroll
    for (int i = 0; i < 16; ++i) {
        int kk = frag_k(i, hlf);
        a[i] = bfbits(zb[(size_t)(tm * 16 + r) * kLAT + kk]);
        b[i] = bfbits(zb[(size_t)(tn * 16 + r) * kLAT + kk]);  // B = z^T : B[k][n] = z[n][k]
    }
    v8f c = {0.f, 0.f, 0.f, 0.f, 0.f, 0.f, 0.f, 0.f};
    c = __builtin_amdgcn_wmma_f32_16x16x32_bf16(false, a, false, b, (short)0, c, false, false);
#pragma unroll
    for (int rr = 0; rr < 8; ++rr) {
        float v = c[rr];
        out[(size_t)(tm * 16 + hlf * 8 + rr) * kN + tn * 16 + r] = 1.f / (1.f + __expf(-v));
    }
}

// ---------------- host launch ----------------
extern "C" void kernel_launch(void* const* d_in, const int* in_sizes, int n_in,
                              void* d_out, int out_size, void* d_ws, size_t ws_size,
                              hipStream_t stream) {
    const float* x       = (const float*)d_in[0];  // [N, 256]
    const float* A       = (const float*)d_in[1];  // [N, N]
    const float* W_heads = (const float*)d_in[2];  // [4, 256, 64]
    const float* a_heads = (const float*)d_in[3];  // [4, 128]
    const float* W_out   = (const float*)d_in[4];  // [256, 32]
    const float* a_out   = (const float*)d_in[5];  // [64]
    float* out = (float*)d_out;                    // [N, N]
    (void)in_sizes; (void)n_in; (void)out_size; (void)ws_size;

    // carve workspace
    uint8_t* p = (uint8_t*)d_ws;
    auto carve = [&](size_t bytes) -> void* {
        void* q = (void*)p;
        p += (bytes + 255) & ~size_t(255);
        return q;
    };
    unsigned short* xbf   = (unsigned short*)carve((size_t)kN * kIN * 2);
    unsigned short* wcbf  = (unsigned short*)carve((size_t)kIN * (kHEADS * kHID) * 2);
    float*          wh    = (float*)carve((size_t)kN * (kHEADS * kHID) * 4);
    unsigned short* whbf  = (unsigned short*)carve((size_t)kN * (kHEADS * kHID) * 2);
    float*          esrc1 = (float*)carve((size_t)kHEADS * kN * 4);
    float*          edst1 = (float*)carve((size_t)kHEADS * kN * 4);
    float*          rmax1 = (float*)carve((size_t)kHEADS * kN * 4);
    float*          rsum1 = (float*)carve((size_t)kHEADS * kN * 4);
    float*          hmat  = (float*)carve((size_t)kN * (kHEADS * kHID) * 4);
    unsigned short* hbf   = (unsigned short*)carve((size_t)kN * (kHEADS * kHID) * 2);
    unsigned short* wobf  = (unsigned short*)carve((size_t)(kHEADS * kHID) * kLAT * 2);
    float*          whz   = (float*)carve((size_t)kN * kLAT * 4);
    unsigned short* whzbf = (unsigned short*)carve((size_t)kN * kLAT * 2);
    float*          esrc2 = (float*)carve((size_t)kN * 4);
    float*          edst2 = (float*)carve((size_t)kN * 4);
    float*          rmax2 = (float*)carve((size_t)kN * 4);
    float*          rsum2 = (float*)carve((size_t)kN * 4);
    float*          z     = (float*)carve((size_t)kN * kLAT * 4);
    unsigned short* zbf   = (unsigned short*)carve((size_t)kN * kLAT * 2);

    auto blocks_for = [](int n) { return (n + 255) / 256; };

    // --- layer 1: Wh = x @ Wc (all heads fused) ---
    f2bf_kernel<<<blocks_for(kN * kIN), 256, 0, stream>>>(x, xbf, kN * kIN);
    pack_wheads_kernel<<<blocks_for(kHEADS * kIN * kHID), 256, 0, stream>>>(W_heads, wcbf);
    {
        int waves = (kN / 16) * ((kHEADS * kHID) / 16);  // 6144
        wmma_gemm_bf16<<<waves / 8, 256, 0, stream>>>(xbf, wcbf, wh, kN, kIN, kHEADS * kHID);
    }
    f2bf_kernel<<<blocks_for(kN * kHEADS * kHID), 256, 0, stream>>>(wh, whbf, kN * kHEADS * kHID);
    edge_logits_kernel<<<blocks_for(kHEADS * kN), 256, 0, stream>>>(
        wh, kHEADS * kHID, kHID, kHEADS, a_heads, esrc1, edst1);

    for (int h = 0; h < kHEADS; ++h) {
        softmax_stats_kernel<<<kN, 256, 0, stream>>>(A, esrc1 + (size_t)h * kN,
                                                     edst1 + (size_t)h * kN,
                                                     rmax1 + (size_t)h * kN,
                                                     rsum1 + (size_t)h * kN);
    }
    for (int h = 0; h < kHEADS; ++h) {
        gat_aggregate_kernel<4><<<(kN / 16) / 8, 256, 0, stream>>>(
            A, esrc1 + (size_t)h * kN, edst1 + (size_t)h * kN,
            rmax1 + (size_t)h * kN, rsum1 + (size_t)h * kN,
            whbf, kHEADS * kHID, h * kHID, hmat, kHEADS * kHID);
    }

    // --- layer 2: Whz = h @ W_out ---
    f2bf_kernel<<<blocks_for(kN * kHEADS * kHID), 256, 0, stream>>>(hmat, hbf, kN * kHEADS * kHID);
    f2bf_kernel<<<blocks_for(kHEADS * kHID * kLAT), 256, 0, stream>>>(W_out, wobf,
                                                                      kHEADS * kHID * kLAT);
    {
        int waves = (kN / 16) * (kLAT / 16);  // 768
        wmma_gemm_bf16<<<waves / 8, 256, 0, stream>>>(hbf, wobf, whz, kN, kHEADS * kHID, kLAT);
    }
    f2bf_kernel<<<blocks_for(kN * kLAT), 256, 0, stream>>>(whz, whzbf, kN * kLAT);
    edge_logits_kernel<<<blocks_for(kN), 256, 0, stream>>>(whz, kLAT, kLAT, 1, a_out, esrc2, edst2);
    softmax_stats_kernel<<<kN, 256, 0, stream>>>(A, esrc2, edst2, rmax2, rsum2);
    gat_aggregate_kernel<2><<<(kN / 16) / 8, 256, 0, stream>>>(
        A, esrc2, edst2, rmax2, rsum2, whzbf, kLAT, 0, z, kLAT);

    // --- output: sigmoid(z @ z^T) ---
    f2bf_kernel<<<blocks_for(kN * kLAT), 256, 0, stream>>>(z, zbf, kN * kLAT);
    {
        int waves = (kN / 16) * (kN / 16);  // 147456
        zzt_sigmoid_kernel<<<waves / 8, 256, 0, stream>>>(zbf, out);
    }
}
